// Block_28235115004006
// MI455X (gfx1250) — matrix operand order repaired and checked by reading, not compiled, not run on hardware
//
#include <hip/hip_runtime.h>
#include <hip/hip_bf16.h>

// ---------------- problem constants ----------------
#define DIM   768
#define HEADS 12
#define HD    64
#define NTOK  197          // N = 14*14 + 1
#define BATCH 64
#define MROWS (BATCH * NTOK)   // 12608 = 197 * 64
#define NPAD  224          // 197 padded to multiple of 32 (WMMA K) and 16
#define HID   3072
#define SCALE 0.125f       // 64^-0.5

typedef __attribute__((ext_vector_type(16))) __bf16    v16bf;
typedef __attribute__((ext_vector_type(8)))  float     v8f;
typedef __attribute__((ext_vector_type(4)))  uint32_t  v4u;
typedef __attribute__((ext_vector_type(8)))  uint32_t  v8u;

// ---------------- WMMA helper (CDNA5 bf16, f32 accumulate) ----------------
__device__ __forceinline__ v8f wmma_bf16(v16bf a, v16bf b, v8f c) {
    // (neg_a, A, neg_b, B, c_mod, C, reuse_a, reuse_b)
    return __builtin_amdgcn_wmma_f32_16x16x32_bf16(false, a, false, b, (short)0, c, false, false);
}

// A-matrix 16x32 bf16 fragment from row-major [16+,ld] tile.
// lane l: row = l&15 ; halves 0..7 -> k0..k0+7, halves 8..15 -> k0+16..k0+23,
// with k0 = (l<16 ? 0 : 8).
__device__ __forceinline__ v16bf load_a_frag(const __bf16* tile, int ld) {
    int lane = threadIdx.x & 31;
    const __bf16* p = tile + (size_t)(lane & 15) * ld + ((lane & 16) ? 8 : 0);
    union { v16bf v; uint4 q[2]; } u;
    u.q[0] = *(const uint4*)(p);
    u.q[1] = *(const uint4*)(p + 16);
    return u.v;
}

// B-matrix 32x16 bf16 fragment from an n-major [16+,ld] tile (row = n, k contiguous).
// lane l: n = l&15 ; halves 0..7 -> k0..k0+7, halves 8..15 -> k0+16..k0+23, k0 = (l<16 ? 0 : 8).
__device__ __forceinline__ v16bf load_b_frag(const __bf16* tile, int ld) {
    int lane = threadIdx.x & 31;
    const __bf16* p = tile + (size_t)(lane & 15) * ld + ((lane & 16) ? 8 : 0);
    union { v16bf v; uint4 q[2]; } u;
    u.q[0] = *(const uint4*)(p);
    u.q[1] = *(const uint4*)(p + 16);
    return u.v;
}

// ---------------- TDM: 2D bf16 tile load, global -> LDS ----------------
// Builds the D# per CDNA5 ISA §8 (group0: count/lds/global/type, group1:
// data_size=2B, tensor dims, tile 32 x tile_rows, row stride in elements)
// and issues tensor_load_to_lds (tracked by TENSORcnt).
__device__ __forceinline__ void tdm_load_tile(uint32_t lds_addr, const __bf16* gsrc,
                                              uint32_t tensor_k, uint32_t tensor_rows,
                                              uint32_t tile_rows) {
    uint64_t ga = (uint64_t)(uintptr_t)gsrc;
    v4u g0;
    g0[0] = 1u;                                            // count = 1 descriptor
    g0[1] = lds_addr;                                      // LDS byte address
    g0[2] = (uint32_t)ga;                                  // global_addr[31:0]
    g0[3] = (uint32_t)((ga >> 32) & 0x01FFFFFFu)           // global_addr[56:32]
          | (2u << 30);                                    // type = 2 ("image")
    v8u g1;
    g1[0] = (1u << 16);                                    // data_size = 1 -> 2 bytes
    g1[1] = (tensor_k & 0xFFFFu) << 16;                    // tensor_dim0[15:0]
    g1[2] = (tensor_k >> 16) | ((tensor_rows & 0xFFFFu) << 16); // dim0 hi | dim1 lo
    g1[3] = (tensor_rows >> 16) | (32u << 16);             // dim1 hi | tile_dim0 = 32
    g1[4] = tile_rows & 0xFFFFu;                           // tile_dim1 (tile_dim2 = 0)
    g1[5] = tensor_k;                                      // tensor_dim0_stride[31:0] (elements)
    g1[6] = 0u;                                            // stride hi / dim1_stride lo
    g1[7] = 0u;
    v4u gz = {0u, 0u, 0u, 0u};                             // groups 2/3 unused (2D)
    asm volatile("tensor_load_to_lds %0, %1, %2, %3"
                 :: "s"(g0), "s"(g1), "s"(gz), "s"(gz)
                 : "memory");
}

// ---------------- fp32 -> bf16 conversion ----------------
__global__ void cvt_f32_bf16(const float* __restrict__ in, __bf16* __restrict__ out, int n) {
    int i = blockIdx.x * blockDim.x + threadIdx.x;
    if (i < n) out[i] = (__bf16)in[i];
}

// ---------------- LayerNorm (768 cols) -> bf16 ----------------
__global__ __launch_bounds__(256)
void ln_bf16(const float* __restrict__ x, const float* __restrict__ w,
             const float* __restrict__ b, __bf16* __restrict__ out) {
    int row = blockIdx.x;
    const float* xr = x + (size_t)row * DIM;
    int t = threadIdx.x;
    float v[3], s = 0.f, s2 = 0.f;
    #pragma unroll
    for (int k = 0; k < 3; ++k) { v[k] = xr[t + 256 * k]; s += v[k]; s2 += v[k] * v[k]; }
    __shared__ float rs[256], rs2[256];
    rs[t] = s; rs2[t] = s2; __syncthreads();
    for (int off = 128; off > 0; off >>= 1) {
        if (t < off) { rs[t] += rs[t + off]; rs2[t] += rs2[t + off]; }
        __syncthreads();
    }
    float mean = rs[0] * (1.f / DIM);
    float var  = rs2[0] * (1.f / DIM) - mean * mean;
    float rstd = rsqrtf(var + 1e-6f);
    #pragma unroll
    for (int k = 0; k < 3; ++k) {
        int c = t + 256 * k;
        out[(size_t)row * DIM + c] = (__bf16)((v[k] - mean) * rstd * w[c] + b[c]);
    }
}

// ---------------- generic WMMA GEMM: C[M,N] = A[M,K] * W[N,K]^T ----------------
// Block tile 64(M) x 256(N), BK=32, 256 threads = 8 waves (2x4), wave tile 32x64.
// Global->LDS staging via TDM (tensor_load_to_lds), double buffered so the next
// tile's DMA overlaps the current tile's 8 WMMAs per wave.
// MODE 0: f32 out (+bias if non-null)   MODE 1: bf16 out, bias+GELU
// MODE 2: f32 out, bias + residual
template<int MODE>
__global__ __launch_bounds__(256)
void gemm_bf16(const __bf16* __restrict__ A, const __bf16* __restrict__ W,
               const float* __restrict__ bias, const float* __restrict__ res,
               void* __restrict__ Cout, int M, int N, int K) {
    __shared__ __align__(16) __bf16 As[2][64 * 32];
    __shared__ __align__(16) __bf16 Bs[2][256 * 32];

    int tid = threadIdx.x;
    int m0 = blockIdx.x * 64;
    int n0 = blockIdx.y * 256;
    int wid = tid >> 5;
    int wm = (wid & 1) * 32;
    int wn = (wid >> 1) * 64;

    const __bf16* Abase = A + (size_t)m0 * K;
    const __bf16* Wbase = W + (size_t)n0 * K;
    int nIter = K >> 5;

    if (wid == 0) {   // TDM ignores EXEC; duplicates (if predicated) are benign
        tdm_load_tile((uint32_t)(uintptr_t)&As[0][0], Abase, (uint32_t)K, (uint32_t)M, 64u);
        tdm_load_tile((uint32_t)(uintptr_t)&Bs[0][0], Wbase, (uint32_t)K, (uint32_t)N, 256u);
    }

    v8f acc[2][4] = {};

    for (int it = 0; it < nIter; ++it) {
        int cur = it & 1;
        if (wid == 0) {
            if (it + 1 < nIter) {
                int kt = (it + 1) << 5;
                tdm_load_tile((uint32_t)(uintptr_t)&As[cur ^ 1][0], Abase + kt,
                              (uint32_t)K, (uint32_t)M, 64u);
                tdm_load_tile((uint32_t)(uintptr_t)&Bs[cur ^ 1][0], Wbase + kt,
                              (uint32_t)K, (uint32_t)N, 256u);
                __builtin_amdgcn_s_wait_tensorcnt(2);   // current buffer's pair done
            } else {
                __builtin_amdgcn_s_wait_tensorcnt(0);
            }
        }
        __syncthreads();

        const __bf16* as = As[cur];
        const __bf16* bs = Bs[cur];
        v16bf a0 = load_a_frag(as + (wm + 0)  * 32, 32);
        v16bf a1 = load_a_frag(as + (wm + 16) * 32, 32);
        #pragma unroll
        for (int tn = 0; tn < 4; ++tn) {
            v16bf bb = load_b_frag(bs + (wn + tn * 16) * 32, 32);
            acc[0][tn] = wmma_bf16(a0, bb, acc[0][tn]);
            acc[1][tn] = wmma_bf16(a1, bb, acc[1][tn]);
        }
        __syncthreads();
    }

    int lane  = tid & 31;
    int colL  = lane & 15;
    int rhalf = (lane >> 4) * 8;
    #pragma unroll
    for (int tm = 0; tm < 2; ++tm)
    #pragma unroll
    for (int tn = 0; tn < 4; ++tn)
    #pragma unroll
    for (int r = 0; r < 8; ++r) {
        int gm = m0 + wm + tm * 16 + r + rhalf;
        int gn = n0 + wn + tn * 16 + colL;
        size_t idx = (size_t)gm * N + gn;
        float v = acc[tm][tn][r];
        if (bias) v += bias[gn];
        if (MODE == 0) {
            ((float*)Cout)[idx] = v;
        } else if (MODE == 1) {
            float g = 0.5f * v * (1.f + erff(v * 0.70710678118654752f));
            ((__bf16*)Cout)[idx] = (__bf16)g;
        } else {
            ((float*)Cout)[idx] = v + res[idx];
        }
    }
}

// ---------------- QKV pack: fp32 C[M,2304] -> bf16 Q,K [bh,224,64], Vt [bh,64,224] ----------------
__global__ __launch_bounds__(256)
void qkv_pack(const float* __restrict__ C, const float* __restrict__ qb,
              const float* __restrict__ vb, __bf16* __restrict__ Q,
              __bf16* __restrict__ Kb, __bf16* __restrict__ Vt) {
    int idx = blockIdx.x * 256 + threadIdx.x;     // < 768*224*64
    int d   = idx & 63;
    int tmp = idx >> 6;
    int i   = tmp % NPAD;
    int bh  = tmp / NPAD;                          // b*12 + h
    int h   = bh % HEADS, b = bh / HEADS;
    int c   = h * HD + d;
    float qv = 0.f, kv = 0.f, vv = 0.f;
    if (i < NTOK) {
        size_t m = (size_t)(b * NTOK + i) * (3 * DIM);
        qv = (C[m + c] + qb[c]) * SCALE;
        kv =  C[m + DIM + c];
        vv =  C[m + 2 * DIM + c] + vb[c];
    }
    size_t base = (size_t)bh * NPAD * HD;
    Q [base + (size_t)i * HD + d]   = (__bf16)qv;
    Kb[base + (size_t)i * HD + d]   = (__bf16)kv;
    Vt[base + (size_t)d * NPAD + i] = (__bf16)vv;
}

// ---------------- attention scores: S[bh,i,j] = Q.K^T + relposbias ----------------
// one 16x16 WMMA tile per wave, K=64 (2 steps), direct global fragment loads
__global__ __launch_bounds__(256)
void attn_scores(const __bf16* __restrict__ Q, const __bf16* __restrict__ Kb,
                 const float* __restrict__ rpb_h, const float* __restrict__ rpb_w,
                 const int* __restrict__ hidx, const int* __restrict__ widx,
                 float* __restrict__ S) {
    int bh = blockIdx.y;
    int h  = bh % HEADS;
    int wid  = threadIdx.x >> 5;
    int tile = blockIdx.x * 8 + wid;               // 14*14 = 196 tiles
    if (tile >= 196) return;                       // wave-uniform exit
    int ti = tile / 14, tj = tile % 14;
    const __bf16* Qb = Q  + (size_t)bh * NPAD * HD;
    const __bf16* KB = Kb + (size_t)bh * NPAD * HD;
    v8f acc = {};
    #pragma unroll
    for (int kk = 0; kk < HD; kk += 32) {
        v16bf a = load_a_frag(Qb + (size_t)(ti * 16) * HD + kk, HD);
        v16bf b = load_b_frag(KB + (size_t)(tj * 16) * HD + kk, HD);
        acc = wmma_bf16(a, b, acc);
    }
    int lane = threadIdx.x & 31;
    int j  = tj * 16 + (lane & 15);
    int ro = (lane >> 4) * 8;
    #pragma unroll
    for (int r = 0; r < 8; ++r) {
        int i = ti * 16 + r + ro;
        float s = acc[r];
        if (i < NTOK && j < NTOK) {
            int e = i * NTOK + j;
            s += rpb_h[hidx[e] * HEADS + h] + rpb_w[widx[e] * HEADS + h];
        }
        S[((size_t)bh * NPAD + i) * NPAD + j] = s;
    }
}

// ---------------- softmax rows -> bf16 P (zero padded) ----------------
__global__ __launch_bounds__(256)
void softmax_rows(const float* __restrict__ S, __bf16* __restrict__ P) {
    int i  = blockIdx.x;           // 0..223
    int bh = blockIdx.y;
    size_t ro = ((size_t)bh * NPAD + i) * NPAD;
    int t = threadIdx.x;
    if (i >= NTOK) {               // padded rows: zero them (block-uniform branch)
        if (t < NPAD) P[ro + t] = (__bf16)0.f;
        return;
    }
    bool valid = t < NTOK;
    float v = valid ? S[ro + t] : -3.4e38f;
    __shared__ float red[256];
    red[t] = v; __syncthreads();
    for (int off = 128; off > 0; off >>= 1) {
        if (t < off) red[t] = fmaxf(red[t], red[t + off]);
        __syncthreads();
    }
    float mx = red[0]; __syncthreads();
    float e = valid ? expf(v - mx) : 0.f;
    red[t] = e; __syncthreads();
    for (int off = 128; off > 0; off >>= 1) {
        if (t < off) red[t] += red[t + off];
        __syncthreads();
    }
    float rsum = 1.f / red[0];
    if (t < NPAD) P[ro + t] = (__bf16)(valid ? e * rsum : 0.f);
}

// ---------------- PV: ctx[m,768] = P[bh,224,224] x V ; writes bf16 packed ----------------
__global__ __launch_bounds__(256)
void attn_pv(const __bf16* __restrict__ P, const __bf16* __restrict__ Vt,
             __bf16* __restrict__ ctx) {
    int bh = blockIdx.y;
    int b = bh / HEADS, h = bh % HEADS;
    int wid  = threadIdx.x >> 5;
    int tile = blockIdx.x * 8 + wid;               // 14 i-tiles * 4 d-tiles = 56
    int ti = tile >> 2, td = tile & 3;
    const __bf16* Pb = P  + (size_t)bh * NPAD * NPAD;
    const __bf16* Vb = Vt + (size_t)bh * HD * NPAD;
    v8f acc = {};
    #pragma unroll
    for (int kk = 0; kk < NPAD; kk += 32) {
        v16bf a = load_a_frag(Pb + (size_t)(ti * 16) * NPAD + kk, NPAD);
        v16bf bfr = load_b_frag(Vb + (size_t)(td * 16) * NPAD + kk, NPAD);
        acc = wmma_bf16(a, bfr, acc);
    }
    int lane = threadIdx.x & 31;
    int d  = td * 16 + (lane & 15);
    int ro = (lane >> 4) * 8;
    #pragma unroll
    for (int r = 0; r < 8; ++r) {
        int i = ti * 16 + r + ro;
        if (i < NTOK)
            ctx[((size_t)(b * NTOK + i)) * DIM + h * HD + d] = (__bf16)acc[r];
    }
}

// ---------------- host launch ----------------
extern "C" void kernel_launch(void* const* d_in, const int* in_sizes, int n_in,
                              void* d_out, int out_size, void* d_ws, size_t ws_size,
                              hipStream_t stream) {
    const float* x       = (const float*)d_in[0];
    const float* norm1_w = (const float*)d_in[1];
    const float* norm1_b = (const float*)d_in[2];
    const float* qkv_w   = (const float*)d_in[3];
    const float* q_bias  = (const float*)d_in[4];
    const float* v_bias  = (const float*)d_in[5];
    const float* proj_w  = (const float*)d_in[6];
    const float* proj_b  = (const float*)d_in[7];
    const float* rpb_h   = (const float*)d_in[8];
    const float* rpb_w   = (const float*)d_in[9];
    const float* norm2_w = (const float*)d_in[10];
    const float* norm2_b = (const float*)d_in[11];
    const float* fc1_w   = (const float*)d_in[12];
    const float* fc1_b   = (const float*)d_in[13];
    const float* fc2_w   = (const float*)d_in[14];
    const float* fc2_b   = (const float*)d_in[15];
    const int*   hidx    = (const int*)d_in[16];
    const int*   widx    = (const int*)d_in[17];
    float* out = (float*)d_out;

    char* ws = (char*)d_ws;
    // workspace layout (bytes), phase-overlapped
    const size_t o_w   = 0;                           // bf16 weights (14.2 MB)
    const size_t o_x   = 14155776;                    // bf16 [M,768]: xln -> ctx2 -> h2
    const size_t o_big = o_x + 19365888;              // f32: C_qkv (116MB) then S (154MB)
    const size_t o_qkv = o_big + 154140672;           // bf16 Q,K,Vt (66MB)
    const size_t o_pm  = o_qkv + 66060288;            // bf16 P (77MB) then m (77.5MB)
    const size_t o_x1  = o_pm + 77463552;             // f32 x1 residual (38.7MB)

    __bf16* wq = (__bf16*)(ws + o_w);
    __bf16* wp = wq + 3 * DIM * DIM;
    __bf16* w1 = wp + DIM * DIM;
    __bf16* w2 = w1 + HID * DIM;
    __bf16* xact = (__bf16*)(ws + o_x);               // reused: xln / ctx2 / h2
    float*  Cqkv = (float*)(ws + o_big);
    float*  S    = (float*)(ws + o_big);
    __bf16* Q  = (__bf16*)(ws + o_qkv);
    __bf16* Kb = Q + (size_t)768 * NPAD * HD;
    __bf16* Vt = Kb + (size_t)768 * NPAD * HD;
    __bf16* P  = (__bf16*)(ws + o_pm);
    __bf16* m  = (__bf16*)(ws + o_pm);
    float*  x1 = (float*)(ws + o_x1);

    // 1) weights -> bf16
    cvt_f32_bf16<<<(3 * DIM * DIM + 255) / 256, 256, 0, stream>>>(qkv_w, wq, 3 * DIM * DIM);
    cvt_f32_bf16<<<(DIM * DIM + 255) / 256, 256, 0, stream>>>(proj_w, wp, DIM * DIM);
    cvt_f32_bf16<<<(HID * DIM + 255) / 256, 256, 0, stream>>>(fc1_w, w1, HID * DIM);
    cvt_f32_bf16<<<(DIM * HID + 255) / 256, 256, 0, stream>>>(fc2_w, w2, DIM * HID);

    // 2) LN1 -> bf16
    ln_bf16<<<MROWS, 256, 0, stream>>>(x, norm1_w, norm1_b, xact);

    // 3) QKV GEMM (f32 out, bias folded in during pack)
    gemm_bf16<0><<<dim3(MROWS / 64, (3 * DIM) / 256), 256, 0, stream>>>(
        xact, wq, nullptr, nullptr, (void*)Cqkv, MROWS, 3 * DIM, DIM);

    // 4) pack into padded per-head buffers
    qkv_pack<<<(768 * NPAD * HD) / 256, 256, 0, stream>>>(Cqkv, q_bias, v_bias, Q, Kb, Vt);

    // 5) scores + rel-pos bias
    attn_scores<<<dim3(25, BATCH * HEADS), 256, 0, stream>>>(Q, Kb, rpb_h, rpb_w, hidx, widx, S);

    // 6) softmax -> bf16 P
    softmax_rows<<<dim3(NPAD, BATCH * HEADS), 256, 0, stream>>>(S, P);

    // 7) PV -> packed bf16 context (overwrites xact; xln is dead)
    attn_pv<<<dim3(7, BATCH * HEADS), 256, 0, stream>>>(P, Vt, xact);

    // 8) proj + residual -> x1 (f32)
    gemm_bf16<2><<<dim3(MROWS / 64, DIM / 256), 256, 0, stream>>>(
        xact, wp, proj_b, x, (void*)x1, MROWS, DIM, DIM);

    // 9) LN2 -> h2 bf16 (reuse xact)
    ln_bf16<<<MROWS, 256, 0, stream>>>(x1, norm2_w, norm2_b, xact);

    // 10) FC1 + GELU -> bf16 m
    gemm_bf16<1><<<dim3(MROWS / 64, HID / 256), 256, 0, stream>>>(
        xact, w1, fc1_b, nullptr, (void*)m, MROWS, HID, DIM);

    // 11) FC2 + residual -> d_out (f32)
    gemm_bf16<2><<<dim3(MROWS / 64, DIM / 256), 256, 0, stream>>>(
        m, w2, fc2_b, x1, (void*)out, MROWS, DIM, HID);
}
